// GroupedQueryAttention_45518063403606
// MI455X (gfx1250) — matrix operand order repaired and checked by reading, not compiled, not run on hardware
//
#include <hip/hip_runtime.h>
#include <hip/hip_bf16.h>

// ---------------------------------------------------------------------------
// CDNA5 (gfx1250, wave32) GQA forward using v_wmma_f32_16x16x32_bf16.
// 1/sqrt(d_k) folded into the Q projection; mask applied in the (coalesced)
// softmax pass instead of the WMMA scores loop.
// ---------------------------------------------------------------------------

typedef __attribute__((ext_vector_type(16))) __bf16 v16bf;
typedef __attribute__((ext_vector_type(8)))  float  v8f;

union Frag16 { v16bf v; unsigned u[8]; };

#define S_LEN   2048
#define D_MODEL 2048
#define NHEADS  32
#define KVDIM   512   // 8 kv heads * 64
#define DK      64

// A-matrix 16x32 bf16 fragment (ISA 7.12.2): lane L -> row M = L%16;
// lanes 0-15 carry K={0..7,16..23}, lanes 16-31 carry K={8..15,24..31};
// each VGPR packs 2 consecutive K (lo = even K).
__device__ inline v16bf load_a_frag(const __bf16* lds, int row0, int stride) {
  const int lane = threadIdx.x & 31;
  const __bf16* p = lds + (size_t)(row0 + (lane & 15)) * stride + ((lane >> 4) * 8);
  Frag16 f;
#pragma unroll
  for (int v = 0; v < 4; ++v) f.u[v]     = *(const unsigned*)(p + 2 * v);
#pragma unroll
  for (int v = 0; v < 4; ++v) f.u[4 + v] = *(const unsigned*)(p + 16 + 2 * v);
  return f.v;
}

// B-matrix 32x16 bf16 fragment: lane L -> col N = L%16; lanes 0-15 K={0..7,16..23},
// lanes 16-31 K={8..15,24..31}; each VGPR packs 2 consecutive K (lo = even K).
// LDS tile must be stored [N][K] so the K-pair is contiguous.
__device__ inline v16bf load_b_frag(const __bf16* lds, int col0, int stride) {
  const int lane = threadIdx.x & 31;
  const __bf16* p = lds + (size_t)(col0 + (lane & 15)) * stride + ((lane >> 4) * 8);
  Frag16 f;
#pragma unroll
  for (int v = 0; v < 8; ++v) f.u[v] = *(const unsigned*)(p + 2 * (v & 3) + 16 * (v >> 2));
  return f.v;
}

__device__ inline v8f wmma_bf16(v16bf a, v16bf b, v8f c) {
  return __builtin_amdgcn_wmma_f32_16x16x32_bf16(
      /*neg_a=*/false, a, /*neg_b=*/false, b,
      /*c_mod=*/(short)0, c, /*reuse_a=*/false, /*reuse_b=*/false);
}

// ---------------------------------------------------------------------------
// Generic tiled GEMM + bias + scale:  C[M,N] = ((A[M,K] @ B[K,N]) + bias) * s
// A is f32 or bf16 (converted to bf16 in LDS), B is f32, C is f32 or bf16.
// 128x128 C-tile per workgroup, 8 waves in a 4x2 grid, K-step 32.
// ---------------------------------------------------------------------------
template <typename AT, bool OUT_BF16>
__global__ __launch_bounds__(256) void gemm_bias_kernel(
    const AT* __restrict__ A, const float* __restrict__ B,
    const float* __restrict__ bias, void* __restrict__ Cout,
    int M, int N, int K, float scale) {
  __shared__ __bf16 Alds[128 * 34];   // [row][k], stride 34 (17-bank, conflict-free)
  __shared__ __bf16 Blds[128 * 34];   // [n][k]   (transposed for B fragments)
  const int t    = threadIdx.x;
  const int wid  = t >> 5, lane = t & 31;
  const int wm   = wid >> 1, wn = wid & 1;
  const int m0   = blockIdx.y * 128, n0 = blockIdx.x * 128;

  v8f acc[2][4] = {};

  for (int k0 = 0; k0 < K; k0 += 32) {
    __syncthreads();
    {  // A tile 128x32 -> LDS (convert to bf16)
      const int r = t >> 3, c = (t & 7) * 4;
#pragma unroll
      for (int p = 0; p < 4; ++p) {
        const int row = r + p * 32;
        const AT* ap = A + (size_t)(m0 + row) * K + k0 + c;
        __bf16* dst = &Alds[row * 34 + c];
        if constexpr (sizeof(AT) == 4) {
          float4 val = *(const float4*)ap;
          dst[0] = (__bf16)val.x; dst[1] = (__bf16)val.y;
          dst[2] = (__bf16)val.z; dst[3] = (__bf16)val.w;
        } else {
          dst[0] = ap[0]; dst[1] = ap[1]; dst[2] = ap[2]; dst[3] = ap[3];
        }
      }
    }
    {  // B tile 32x128 -> LDS transposed [n][k]
      const int kk0 = t >> 5, c = (t & 31) * 4;
#pragma unroll
      for (int p = 0; p < 4; ++p) {
        const int kk = kk0 + p * 8;
        const float* bp = B + (size_t)(k0 + kk) * N + n0 + c;
        float4 val = *(const float4*)bp;
        Blds[(c + 0) * 34 + kk] = (__bf16)val.x;
        Blds[(c + 1) * 34 + kk] = (__bf16)val.y;
        Blds[(c + 2) * 34 + kk] = (__bf16)val.z;
        Blds[(c + 3) * 34 + kk] = (__bf16)val.w;
      }
    }
    __syncthreads();

    v16bf bfr[4];
#pragma unroll
    for (int nt = 0; nt < 4; ++nt) bfr[nt] = load_b_frag(Blds, wn * 64 + nt * 16, 34);
#pragma unroll
    for (int mt = 0; mt < 2; ++mt) {
      v16bf a = load_a_frag(Alds, wm * 32 + mt * 16, 34);
#pragma unroll
      for (int nt = 0; nt < 4; ++nt) acc[mt][nt] = wmma_bf16(a, bfr[nt], acc[mt][nt]);
    }
  }

  // C/D layout: VGPR r -> M = r + 8*lane[4]; lane[3:0] -> N.
#pragma unroll
  for (int mt = 0; mt < 2; ++mt)
#pragma unroll
    for (int nt = 0; nt < 4; ++nt)
#pragma unroll
      for (int r = 0; r < 8; ++r) {
        const int grow = m0 + wm * 32 + mt * 16 + r + (lane >> 4) * 8;
        const int gcol = n0 + wn * 64 + nt * 16 + (lane & 15);
        const float val = (acc[mt][nt][r] + bias[gcol]) * scale;
        if constexpr (OUT_BF16)
          ((__bf16*)Cout)[(size_t)grow * N + gcol] = (__bf16)val;
        else
          ((float*)Cout)[(size_t)grow * N + gcol] = val;
      }
}

// ---------------------------------------------------------------------------
// Scores: raw S = (Q_h/8) @ K_g^T, one (head, 64-row) block per workgroup.
// Pure WMMA + accumulator stores; masking deferred to the softmax pass.
// ---------------------------------------------------------------------------
__global__ __launch_bounds__(256) void scores_kernel(
    const __bf16* __restrict__ Qbf, const __bf16* __restrict__ Kbf,
    float* __restrict__ attnp) {
  const int qb = blockIdx.x, h = blockIdx.y, g = h >> 2;  // 4 query heads per kv head
  __shared__ __bf16 Qlds[64 * 72];    // [qrow][d], stride 72 (36-bank, conflict-free)
  __shared__ __bf16 Klds[128 * 72];   // [kv][d] == B-operand [n][k] layout

  {  // Q tile 64x64 (already bf16): 16 elements/thread, 16B vector moves
    const int row = threadIdx.x >> 2, cb = (threadIdx.x & 3) * 16;
    const __bf16* qp = Qbf + (size_t)(qb * 64 + row) * D_MODEL + h * DK + cb;
    *(uint4*)&Qlds[row * 72 + cb]     = ((const uint4*)qp)[0];
    *(uint4*)&Qlds[row * 72 + cb + 8] = ((const uint4*)qp)[1];
  }
  __syncthreads();

  const int wid = threadIdx.x >> 5, lane = threadIdx.x & 31;

  // Hoist all 8 Q fragments (invariant across the KV loop) into registers.
  v16bf qfr[2][4];
#pragma unroll
  for (int d0 = 0; d0 < 2; ++d0)
#pragma unroll
    for (int mt = 0; mt < 4; ++mt)
      qfr[d0][mt] = load_a_frag(Qlds + d0 * 32, mt * 16, 72);

  for (int kv0 = 0; kv0 < S_LEN; kv0 += 128) {
    __syncthreads();
    {  // K tile 128x64 bf16
      const int krow = threadIdx.x >> 1, cb = (threadIdx.x & 1) * 32;
      const __bf16* kp = Kbf + (size_t)(kv0 + krow) * KVDIM + g * DK + cb;
#pragma unroll
      for (int j = 0; j < 4; ++j)
        *(uint4*)&Klds[krow * 72 + cb + j * 8] = ((const uint4*)kp)[j];
    }
    __syncthreads();

    v8f acc[4] = {};  // this wave: 64 rows x 16 cols (col tile = wid)
#pragma unroll
    for (int d0 = 0; d0 < 2; ++d0) {
      v16bf b = load_b_frag(Klds + d0 * 32, wid * 16, 72);
#pragma unroll
      for (int mt = 0; mt < 4; ++mt) acc[mt] = wmma_bf16(qfr[d0][mt], b, acc[mt]);
    }

#pragma unroll
    for (int mt = 0; mt < 4; ++mt)
#pragma unroll
      for (int r = 0; r < 8; ++r) {
        const int qrow = qb * 64 + mt * 16 + r + (lane >> 4) * 8;
        const int col  = kv0 + wid * 16 + (lane & 15);
        attnp[((size_t)h * S_LEN + qrow) * S_LEN + col] = acc[mt][r];
      }
  }
}

// ---------------------------------------------------------------------------
// In-place masked row softmax over 2048 columns, one (head,row) per
// workgroup. Mask bytes stream contiguously (fully coalesced; the 4 MB mask
// stays L2-resident across the 32 heads that reuse each row).
// ---------------------------------------------------------------------------
__global__ __launch_bounds__(256) void softmax_kernel(
    float* __restrict__ attnp, const unsigned char* __restrict__ mask) {
  const int row  = blockIdx.x;            // h*S + qrow
  const int qrow = row & (S_LEN - 1);
  float* base = attnp + (size_t)row * S_LEN;
  const unsigned char* mbase = mask + (size_t)qrow * S_LEN;
  const int t = threadIdx.x;
  __shared__ float red[256];

  float r[8];
  float m = -1e30f;
#pragma unroll
  for (int i = 0; i < 8; ++i) {
    float x = base[t + i * 256];
    if (!mbase[t + i * 256]) x = -1e9f;   // reference: where(mask, s, -1e9)
    r[i] = x;
    m = fmaxf(m, x);
  }
  red[t] = m; __syncthreads();
  for (int s = 128; s > 0; s >>= 1) { if (t < s) red[t] = fmaxf(red[t], red[t + s]); __syncthreads(); }
  m = red[0]; __syncthreads();

  float sum = 0.f;
#pragma unroll
  for (int i = 0; i < 8; ++i) { r[i] = __expf(r[i] - m); sum += r[i]; }
  red[t] = sum; __syncthreads();
  for (int s = 128; s > 0; s >>= 1) { if (t < s) red[t] += red[t + s]; __syncthreads(); }
  const float inv = 1.0f / red[0];
#pragma unroll
  for (int i = 0; i < 8; ++i) base[t + i * 256] = r[i] * inv;
}

// ---------------------------------------------------------------------------
// attn @ V: one (head, 64-row) block per workgroup -> bf16 head outputs
// laid out [s, h*64+d]. K-step 64 (two WMMA k-substeps) to amortize barriers.
// ---------------------------------------------------------------------------
__global__ __launch_bounds__(256) void av_kernel(
    const float* __restrict__ attnp, const __bf16* __restrict__ Vbf,
    __bf16* __restrict__ Obf) {
  const int qb = blockIdx.x, h = blockIdx.y, g = h >> 2;
  __shared__ __bf16 Alds[64 * 72];  // attn tile [qrow][kv0..63]
  __shared__ __bf16 Vlds[64 * 72];  // V tile transposed [d][kv0..63] (B operand)
  const int wid = threadIdx.x >> 5, lane = threadIdx.x & 31;
  const int mt = wid >> 1, ntb = (wid & 1) * 2;  // wave: 16 rows x 32 d-cols

  v8f acc[2] = {};
  const float* ap_base = attnp + ((size_t)h * S_LEN + qb * 64) * S_LEN;

  for (int k0 = 0; k0 < S_LEN; k0 += 64) {
    __syncthreads();
    {  // attn 64x64 f32 -> bf16 LDS
      const int row = threadIdx.x >> 2, cb = (threadIdx.x & 3) * 16;
      const float* ap = ap_base + (size_t)row * S_LEN + k0 + cb;
      __bf16* dst = &Alds[row * 72 + cb];
#pragma unroll
      for (int q4 = 0; q4 < 4; ++q4) {
        float4 u = ((const float4*)ap)[q4];
        dst[q4 * 4 + 0] = (__bf16)u.x; dst[q4 * 4 + 1] = (__bf16)u.y;
        dst[q4 * 4 + 2] = (__bf16)u.z; dst[q4 * 4 + 3] = (__bf16)u.w;
      }
    }
    {  // V 64x64 bf16 -> transposed [d][kv]
      const int kv = threadIdx.x >> 2, db = (threadIdx.x & 3) * 16;
      const __bf16* vp = Vbf + (size_t)(k0 + kv) * KVDIM + g * DK + db;
#pragma unroll
      for (int i = 0; i < 16; ++i) Vlds[(db + i) * 72 + kv] = vp[i];
    }
    __syncthreads();

#pragma unroll
    for (int d0 = 0; d0 < 2; ++d0) {  // two K=32 substeps
      v16bf a = load_a_frag(Alds + d0 * 32, mt * 16, 72);
#pragma unroll
      for (int j = 0; j < 2; ++j) {
        v16bf b = load_b_frag(Vlds + d0 * 32, (ntb + j) * 16, 72);
        acc[j] = wmma_bf16(a, b, acc[j]);
      }
    }
  }

#pragma unroll
  for (int j = 0; j < 2; ++j)
#pragma unroll
    for (int r = 0; r < 8; ++r) {
      const int qrow = qb * 64 + mt * 16 + r + (lane >> 4) * 8;
      const int dcol = (ntb + j) * 16 + (lane & 15);
      Obf[(size_t)qrow * D_MODEL + h * DK + dcol] = (__bf16)acc[j][r];
    }
}

// ---------------------------------------------------------------------------
extern "C" void kernel_launch(void* const* d_in, const int* in_sizes, int n_in,
                              void* d_out, int out_size, void* d_ws, size_t ws_size,
                              hipStream_t stream) {
  (void)in_sizes; (void)n_in; (void)out_size; (void)ws_size;
  const float* q  = (const float*)d_in[0];
  const float* k  = (const float*)d_in[1];
  const float* v  = (const float*)d_in[2];
  const unsigned char* mask = (const unsigned char*)d_in[3];
  const float* wq = (const float*)d_in[4];
  const float* bq = (const float*)d_in[5];
  const float* wk = (const float*)d_in[6];
  const float* bk = (const float*)d_in[7];
  const float* wv = (const float*)d_in[8];
  const float* bv = (const float*)d_in[9];
  const float* wo = (const float*)d_in[10];
  const float* bo = (const float*)d_in[11];

  // workspace: bf16 intermediates (~20 MB)
  __bf16* Qbf = (__bf16*)d_ws;                       // [2048, 2048]  (pre-scaled by 1/8)
  __bf16* Kbf = Qbf + (size_t)S_LEN * D_MODEL;       // [2048, 512]
  __bf16* Vbf = Kbf + (size_t)S_LEN * KVDIM;         // [2048, 512]
  __bf16* Obf = Vbf + (size_t)S_LEN * KVDIM;         // [2048, 2048]

  float* outp  = (float*)d_out;                      // out: 2048*2048 f32
  float* attnp = outp + (size_t)S_LEN * D_MODEL;     // attn: 32*2048*2048 f32

  const dim3 blk(256);
  gemm_bias_kernel<float, true><<<dim3(16, 16), blk, 0, stream>>>(q, wq, bq, Qbf, S_LEN, D_MODEL, D_MODEL, 0.125f);
  gemm_bias_kernel<float, true><<<dim3(4, 16),  blk, 0, stream>>>(k, wk, bk, Kbf, S_LEN, KVDIM, D_MODEL, 1.0f);
  gemm_bias_kernel<float, true><<<dim3(4, 16),  blk, 0, stream>>>(v, wv, bv, Vbf, S_LEN, KVDIM, D_MODEL, 1.0f);
  scores_kernel<<<dim3(32, 32), blk, 0, stream>>>(Qbf, Kbf, attnp);
  softmax_kernel<<<dim3(NHEADS * S_LEN), blk, 0, stream>>>(attnp, mask);
  av_kernel<<<dim3(32, 32), blk, 0, stream>>>(attnp, Vbf, Obf);
  gemm_bias_kernel<__bf16, false><<<dim3(16, 16), blk, 0, stream>>>(Obf, wo, bo, outp, S_LEN, D_MODEL, D_MODEL, 1.0f);
}
